// Decoder_13572096655402
// MI455X (gfx1250) — matrix operand order repaired and checked by reading, not compiled, not run on hardware
//
#include <hip/hip_runtime.h>
#include <hip/hip_bf16.h>

// ---------------- problem constants ----------------
#define BATCH   512
#define HIDDEN  1024
#define OUT_DIM 64
#define SEQ     256
#define K0      1088   // 64 (x) + 1024 (h0)
#define K1      2048   // 1024 (h0) + 1024 (h1)
#define G4H     4096   // 4*HIDDEN gate rows
#define BM      128    // batch rows per workgroup (8 waves x 16)

// LDS tile row stride (elements) : 32 data + 8 pad, keeps 16B alignment
#define LSTR 40

typedef __attribute__((ext_vector_type(16))) __bf16 v16bf;
typedef __attribute__((ext_vector_type(8)))  float  v8f;

// float -> bf16, round-to-nearest-even
__device__ __forceinline__ unsigned short f2bf(float f) {
  union { float f; unsigned int u; } c; c.f = f;
  unsigned int u = c.u;
  u += 0x7fffu + ((u >> 16) & 1u);
  return (unsigned short)(u >> 16);
}

__device__ __forceinline__ float sigmoidf_(float x) {
  return 1.0f / (1.0f + __expf(-x));
}
__device__ __forceinline__ float tanhf_(float x) {
  x = fminf(fmaxf(x, -15.0f), 15.0f);
  float e = __expf(2.0f * x);
  return (e - 1.0f) / (e + 1.0f);
}

// A fragment (16x32 bf16, MxK). Per ISA 7.12.2:
// lanes 0-15 : M=lane,     K=0..7 in v0..3,  K=16..23 in v4..7
// lanes 16-31: M=lane-16,  K=8..15 in v0..3, K=24..31 in v4..7
__device__ __forceinline__ v16bf load_frag_a(const unsigned short* lds, int lane) {
  int row = lane & 15;
  int kb  = (lane >> 4) << 3;               // 0 or 8
  const unsigned short* p = lds + row * LSTR + kb;
  union { v16bf v; uint4 q[2]; } u;
  u.q[0] = *(const uint4*)(p);              // K = kb .. kb+7
  u.q[1] = *(const uint4*)(p + 16);         // K = kb+16 .. kb+23
  return u.v;
}

// B fragment (32x16 bf16, KxN). Per ISA layout:
// lanes 0-15 : N=lane, K=0..7,16..23 ; lanes 16-31: N=lane-16, K=8..15,24..31
__device__ __forceinline__ v16bf load_frag_b(const unsigned short* lds, int lane) {
  int n  = lane & 15;
  int kb = (lane >> 4) << 3;                // 0 or 8
  const unsigned short* p = lds + n * LSTR + kb;
  union { v16bf v; uint4 q[2]; } u;
  u.q[0] = *(const uint4*)(p);
  u.q[1] = *(const uint4*)(p + 16);
  return u.v;
}

// ---------------- fused LSTM layer: gates GEMM + cell ----------------
// gates = A[B,Kdim] @ W[4H,Kdim]^T + bias ; cell update; h_new (bf16) scattered
// to two destination activation buffers. BM=128 batch rows x 16 hidden x 4 gates
// per workgroup; double-buffered LDS staging, one barrier per K-chunk.
__global__ __launch_bounds__(256)
void lstm_layer_kernel(const unsigned short* __restrict__ A,
                       const unsigned short* __restrict__ W,
                       const float* __restrict__ bias,
                       float* __restrict__ cbuf,
                       unsigned short* __restrict__ hdst0, int hstride0, int hoff0,
                       unsigned short* __restrict__ hdst1, int hstride1, int hoff1,
                       int Kdim)
{
  __shared__ __align__(16) unsigned short Atile[2][BM * LSTR];
  __shared__ __align__(16) unsigned short Btile[2][64 * LSTR];

  const int j0   = blockIdx.x << 4;   // hidden unit block (16 wide)
  const int r0   = blockIdx.y * BM;   // batch block (128 rows)
  const int tid  = threadIdx.x;
  const int lane = tid & 31;
  const int wv   = tid >> 5;          // wave 0..7 -> 16-row batch sub-block

  v8f acc[4];
  for (int g = 0; g < 4; ++g)
    for (int e = 0; e < 8; ++e) acc[g][e] = 0.0f;

  // staging assignments
  const int aRow = tid >> 1;                  // 0..127
  const int aSeg = (tid & 1) << 4;            // 0 or 16 elems (32B)
  const int bRow = (tid & 127) >> 1;          // 0..63 (threads 0..127 stage B)
  const int bSeg = (tid & 1) << 4;
  const int gsub = bRow >> 4;                 // gate of this B row
  const int nsub = bRow & 15;
  const int wrow = gsub * HIDDEN + j0 + nsub; // weight row for B staging
  const bool doB = (tid < 128);

  // ---- prologue: stage chunk 0 into buffer 0 ----
  {
    const uint4* sa = (const uint4*)(A + (size_t)(r0 + aRow) * Kdim + aSeg);
    uint4* da = (uint4*)(&Atile[0][aRow * LSTR + aSeg]);
    da[0] = sa[0]; da[1] = sa[1];
    if (doB) {
      const uint4* sb = (const uint4*)(W + (size_t)wrow * Kdim + bSeg);
      uint4* db = (uint4*)(&Btile[0][bRow * LSTR + bSeg]);
      db[0] = sb[0]; db[1] = sb[1];
    }
  }
  __syncthreads();

  int buf = 0;
  for (int k0 = 0; k0 < Kdim; k0 += 32) {
    const int nbuf = buf ^ 1;
    // stage next chunk into the other buffer (overlaps with WMMA below)
    if (k0 + 32 < Kdim) {
      const int kn = k0 + 32;
      const uint4* sa = (const uint4*)(A + (size_t)(r0 + aRow) * Kdim + kn + aSeg);
      uint4* da = (uint4*)(&Atile[nbuf][aRow * LSTR + aSeg]);
      da[0] = sa[0]; da[1] = sa[1];
      if (doB) {
        const uint4* sb = (const uint4*)(W + (size_t)wrow * Kdim + kn + bSeg);
        uint4* db = (uint4*)(&Btile[nbuf][bRow * LSTR + bSeg]);
        db[0] = sb[0]; db[1] = sb[1];
        // pull the chunk after that toward the WGP (global_prefetch_b8)
        if (k0 + 64 < Kdim)
          __builtin_prefetch((const void*)(W + (size_t)wrow * Kdim + k0 + 64 + bSeg), 0, 3);
      }
    }

    v16bf af = load_frag_a(&Atile[buf][(wv << 4) * LSTR], lane);
    #pragma unroll
    for (int g = 0; g < 4; ++g) {
      v16bf bfr = load_frag_b(&Btile[buf][(g << 4) * LSTR], lane);
      acc[g] = __builtin_amdgcn_wmma_f32_16x16x32_bf16(
          false, af, false, bfr, (short)0, acc[g], false, false);
    }
    __syncthreads();
    buf = nbuf;
  }

  // ---- LSTM cell (all 4 gates for (row,col) live in the same lane) ----
  const int col = lane & 15;
  const int rb  = (lane >> 4) << 3;   // 0 or 8
  const float b_i = bias[0 * HIDDEN + j0 + col];
  const float b_f = bias[1 * HIDDEN + j0 + col];
  const float b_g = bias[2 * HIDDEN + j0 + col];
  const float b_o = bias[3 * HIDDEN + j0 + col];

  #pragma unroll
  for (int v = 0; v < 8; ++v) {
    const int r  = r0 + (wv << 4) + rb + v;
    const float gi = sigmoidf_(acc[0][v] + b_i);
    const float gf = sigmoidf_(acc[1][v] + b_f);
    const float gg = tanhf_   (acc[2][v] + b_g);
    const float go = sigmoidf_(acc[3][v] + b_o);
    const int cidx = r * HIDDEN + j0 + col;
    const float c_new = gf * cbuf[cidx] + gi * gg;
    cbuf[cidx] = c_new;
    const unsigned short hb = f2bf(go * tanhf_(c_new));
    hdst0[(size_t)r * hstride0 + hoff0 + j0 + col] = hb;
    hdst1[(size_t)r * hstride1 + hoff1 + j0 + col] = hb;
  }
}

// ---------------- fc_out: out = h1 @ Wout^T + b ; also feed x back ----------------
__global__ __launch_bounds__(128)
void fcout_kernel(const unsigned short* __restrict__ A,   // A1 buffer, h1 at col offset 1024
                  const unsigned short* __restrict__ W,   // [64,1024] bf16
                  const float* __restrict__ bias,          // [64]
                  float* __restrict__ out,                 // [B, SEQ, 64]
                  unsigned short* __restrict__ xdst,       // A0 next, x region
                  int t)
{
  __shared__ __align__(16) unsigned short Atile[64 * LSTR];
  __shared__ __align__(16) unsigned short Btile[64 * LSTR];

  const int r0   = blockIdx.x << 6;
  const int tid  = threadIdx.x;
  const int lane = tid & 31;
  const int wv   = tid >> 5;

  v8f acc[4];
  for (int g = 0; g < 4; ++g)
    for (int e = 0; e < 8; ++e) acc[g][e] = 0.0f;

  const int loadRow = tid >> 1;
  const int loadSeg = (tid & 1) << 4;

  for (int k0 = 0; k0 < HIDDEN; k0 += 32) {
    {
      const uint4* sa = (const uint4*)(A + (size_t)(r0 + loadRow) * K1 + HIDDEN + k0 + loadSeg);
      uint4* da = (uint4*)(Atile + loadRow * LSTR + loadSeg);
      da[0] = sa[0]; da[1] = sa[1];
      const uint4* sb = (const uint4*)(W + (size_t)loadRow * HIDDEN + k0 + loadSeg);
      uint4* db = (uint4*)(Btile + loadRow * LSTR + loadSeg);
      db[0] = sb[0]; db[1] = sb[1];
    }
    __syncthreads();

    v16bf af = load_frag_a(Atile + (wv << 4) * LSTR, lane);
    #pragma unroll
    for (int g = 0; g < 4; ++g) {
      v16bf bfr = load_frag_b(Btile + (g << 4) * LSTR, lane);
      acc[g] = __builtin_amdgcn_wmma_f32_16x16x32_bf16(
          false, af, false, bfr, (short)0, acc[g], false, false);
    }
    __syncthreads();
  }

  const int col = lane & 15;
  const int rb  = (lane >> 4) << 3;
  #pragma unroll
  for (int g = 0; g < 4; ++g) {
    const int n = (g << 4) + col;
    const float bb = bias[n];
    #pragma unroll
    for (int v = 0; v < 8; ++v) {
      const int r = r0 + (wv << 4) + rb + v;
      const float val = acc[g][v] + bb;
      out[(size_t)r * (SEQ * OUT_DIM) + t * OUT_DIM + n] = val;
      xdst[(size_t)r * K0 + n] = f2bf(val);
    }
  }
}

// ---------------- one-time prep kernels ----------------
// dst[r, 0:ka] = a[r,:] ; dst[r, ka:ka+kb] = b[r,:]  (fp32 -> bf16)
__global__ void convcat_kernel(unsigned short* __restrict__ dst,
                               const float* __restrict__ a,
                               const float* __restrict__ b,
                               int rows, int ka, int kb) {
  const int K = ka + kb;
  const int total = rows * K;
  for (int idx = blockIdx.x * blockDim.x + threadIdx.x; idx < total;
       idx += gridDim.x * blockDim.x) {
    const int r = idx / K;
    const int c = idx - r * K;
    const float v = (c < ka) ? a[(size_t)r * ka + c] : b[(size_t)r * kb + (c - ka)];
    dst[idx] = f2bf(v);
  }
}

__global__ void bias_sum_kernel(float* __restrict__ dst,
                                const float* __restrict__ a,
                                const float* __restrict__ b, int n) {
  const int i = blockIdx.x * blockDim.x + threadIdx.x;
  if (i < n) dst[i] = a[i] + b[i];
}

// init_flat = concat(z_style,z_skill) @ fc_init_w^T + fc_init_b  (raw reshape to h0/h1)
__global__ void init_state_kernel(const float* __restrict__ zs,
                                  const float* __restrict__ zk,
                                  const float* __restrict__ w,
                                  const float* __restrict__ bsrc,
                                  unsigned short* __restrict__ A0cur,
                                  unsigned short* __restrict__ A1cur) {
  const int p = blockIdx.x * blockDim.x + threadIdx.x;
  if (p >= 2 * BATCH * HIDDEN) return;
  const int b = p >> 11;          // init_flat row
  const int k = p & 2047;         // init_flat col
  float accv = bsrc[k];
  const float* wr = w + (size_t)k * 256;
  const float* z0 = zs + (size_t)b * 128;
  const float* z1 = zk + (size_t)b * 128;
  #pragma unroll 4
  for (int q = 0; q < 128; ++q) accv += z0[q] * wr[q];
  #pragma unroll 4
  for (int q = 0; q < 128; ++q) accv += z1[q] * wr[128 + q];
  const unsigned short hb = f2bf(accv);
  if (p < BATCH * HIDDEN) {               // h0 (first half of raw flat)
    const int bb = p >> 10, h = p & 1023;
    A0cur[(size_t)bb * K0 + 64 + h] = hb;
  } else {                                 // h1 (second half)
    const int p2 = p - BATCH * HIDDEN;
    const int bb = p2 >> 10, h = p2 & 1023;
    A1cur[(size_t)bb * K1 + HIDDEN + h] = hb;
  }
}

__global__ void zero_state_kernel(float* __restrict__ c0, float* __restrict__ c1,
                                  unsigned short* __restrict__ A0cur) {
  const int i = blockIdx.x * blockDim.x + threadIdx.x;
  if (i < BATCH * HIDDEN) { c0[i] = 0.0f; c1[i] = 0.0f; }
  if (i < BATCH * OUT_DIM) {
    const int b = i >> 6, n = i & 63;
    A0cur[(size_t)b * K0 + n] = 0;   // x0 = zeros
  }
}

// ---------------- host side ----------------
extern "C" void kernel_launch(void* const* d_in, const int* in_sizes, int n_in,
                              void* d_out, int out_size, void* d_ws, size_t ws_size,
                              hipStream_t stream) {
  (void)in_sizes; (void)n_in; (void)out_size; (void)ws_size;
  const float* z_style   = (const float*)d_in[0];
  const float* z_skill   = (const float*)d_in[1];
  const float* fc_init_w = (const float*)d_in[2];
  const float* fc_init_b = (const float*)d_in[3];
  const float* W_ih0     = (const float*)d_in[4];
  const float* W_hh0     = (const float*)d_in[5];
  const float* b_ih0     = (const float*)d_in[6];
  const float* b_hh0     = (const float*)d_in[7];
  const float* W_ih1     = (const float*)d_in[8];
  const float* W_hh1     = (const float*)d_in[9];
  const float* b_ih1     = (const float*)d_in[10];
  const float* b_hh1     = (const float*)d_in[11];
  const float* fc_out_w  = (const float*)d_in[12];
  const float* fc_out_b  = (const float*)d_in[13];
  float* out = (float*)d_out;

  char* ws = (char*)d_ws;
  size_t off = 0;
  auto alloc = [&](size_t bytes) -> void* {
    off = (off + 255) & ~(size_t)255;
    void* p = ws + off;
    off += bytes;
    return p;
  };

  unsigned short* Wc0   = (unsigned short*)alloc((size_t)G4H * K0 * 2);
  unsigned short* Wc1   = (unsigned short*)alloc((size_t)G4H * K1 * 2);
  unsigned short* Wout  = (unsigned short*)alloc((size_t)OUT_DIM * HIDDEN * 2);
  float*          bias0 = (float*)alloc((size_t)G4H * 4);
  float*          bias1 = (float*)alloc((size_t)G4H * 4);
  unsigned short* A0[2];
  A0[0] = (unsigned short*)alloc((size_t)BATCH * K0 * 2);
  A0[1] = (unsigned short*)alloc((size_t)BATCH * K0 * 2);
  unsigned short* A1[2];
  A1[0] = (unsigned short*)alloc((size_t)BATCH * K1 * 2);
  A1[1] = (unsigned short*)alloc((size_t)BATCH * K1 * 2);
  float* c0 = (float*)alloc((size_t)BATCH * HIDDEN * 4);
  float* c1 = (float*)alloc((size_t)BATCH * HIDDEN * 4);

  // one-time prep (recomputed every call -> deterministic)
  convcat_kernel<<<2048, 256, 0, stream>>>(Wc0, W_ih0, W_hh0, G4H, OUT_DIM, HIDDEN);
  convcat_kernel<<<4096, 256, 0, stream>>>(Wc1, W_ih1, W_hh1, G4H, HIDDEN, HIDDEN);
  convcat_kernel<<<256, 256, 0, stream>>>(Wout, fc_out_w, fc_out_w, OUT_DIM, HIDDEN, 0);
  bias_sum_kernel<<<16, 256, 0, stream>>>(bias0, b_ih0, b_hh0, G4H);
  bias_sum_kernel<<<16, 256, 0, stream>>>(bias1, b_ih1, b_hh1, G4H);
  init_state_kernel<<<(2 * BATCH * HIDDEN + 255) / 256, 256, 0, stream>>>(
      z_style, z_skill, fc_init_w, fc_init_b, A0[0], A1[0]);
  zero_state_kernel<<<(BATCH * HIDDEN + 255) / 256, 256, 0, stream>>>(c0, c1, A0[0]);

  const dim3 lgrid(HIDDEN / 16, BATCH / BM);   // 64 x 4 workgroups, 256 thr each
  for (int t = 0; t < SEQ; ++t) {
    const int cur = t & 1;
    const int nxt = cur ^ 1;
    // layer 0: reads A0[cur]=[x|h0]; writes h0_new -> A0[nxt].h and A1[cur].h0
    lstm_layer_kernel<<<lgrid, 256, 0, stream>>>(
        A0[cur], Wc0, bias0, c0,
        A0[nxt], K0, OUT_DIM,
        A1[cur], K1, 0,
        K0);
    // layer 1: reads A1[cur]=[h0_new|h1]; writes h1_new -> A1[nxt].h1
    lstm_layer_kernel<<<lgrid, 256, 0, stream>>>(
        A1[cur], Wc1, bias1, c1,
        A1[nxt], K1, HIDDEN,
        A1[nxt], K1, HIDDEN,
        K1);
    // fc_out: reads h1_new from A1[nxt]; writes out[b,t,:] fp32 and x -> A0[nxt]
    fcout_kernel<<<BATCH / 64, 128, 0, stream>>>(
        A1[nxt], Wout, fc_out_b, out, A0[nxt], t);
  }
}
